// NAFBlockHyper_11862699672008
// MI455X (gfx1250) — matrix operand order repaired and checked by reading, not compiled, not run on hardware
//
#include <hip/hip_runtime.h>

// ---------------------------------------------------------------------------
// NAFNet block (per-sample hyper weights) for MI455X / gfx1250, wave32.
// All 1x1 convs (K=32) run on v_wmma_f32_16x16x32_bf16 (bf16 A/B, f32 acc).
// Folds: biases -> WMMA C-operand; LN gain -> A columns; LN shift -> bias;
//        SE scale -> w3 columns. x1 kept in packed per-pixel bf16 layout so
//        stage2's B-fragment is a single 32B vector load.
// ---------------------------------------------------------------------------

typedef __attribute__((ext_vector_type(16))) __bf16 bf16x16;
typedef __attribute__((ext_vector_type(8)))  __bf16 bf16x8;
typedef __attribute__((ext_vector_type(8)))  float  f32x8;

constexpr int Cc  = 32;      // base channels
constexpr int DW  = 64;      // dw = ffn = 2*c
constexpr int Hh  = 256, Ww = 256, HW = Hh * Ww;
constexpr int Bn  = 16;

constexpr int TILE_W = 32, TILE_H = 8;              // stage1 output tile
constexpr int HALO_W = TILE_W + 2, HALO_H = TILE_H + 2;
constexpr int HALO_PX = HALO_W * HALO_H;            // 340
constexpr int NGROUP  = (HALO_PX + 15) / 16;        // 22
constexpr int USTRIDE = NGROUP * 16;                // 352 (padded)
constexpr int TILES_PER_B = (Ww / TILE_W) * (Hh / TILE_H); // 256

static __device__ inline f32x8 wmma_bf16f32(const bf16x16 a, const bf16x16 b, const f32x8 c) {
    // D(16x16,f32) = A(16x32,bf16) * B(32x16,bf16) + C
    return __builtin_amdgcn_wmma_f32_16x16x32_bf16(false, a, false, b, (short)0, c, false, false);
}

// A-fragment (16-bit A 16x32): lane holds row M = lane&15,
// K = 16h+j (contiguous, matches B fragments), h = lane>>4.
static __device__ inline void build_afrag(const float* Wlds, int mtile, int lane, bf16x16& a) {
    const int h = (lane >> 4) & 1, mrow = lane & 15;
    const float* row = Wlds + (mtile * 16 + mrow) * Cc;
#pragma unroll
    for (int j = 0; j < 16; ++j) {
        const int k = 16 * h + j;
        a[j] = (__bf16)row[k];
    }
}

// C-fragment carrying a per-output-channel bias, broadcast across columns:
// C/D layout: VGPR r <-> row M = 8h+r, all 16 columns identical.
static __device__ inline f32x8 build_cbias(const float* blds, int mtile, int lane) {
    const int h = (lane >> 4) & 1;
    f32x8 c;
#pragma unroll
    for (int r = 0; r < 8; ++r) c[r] = blds[mtile * 16 + 8 * h + r];
    return c;
}

// ---------------------------------------------------------------------------
// Stage 1: LN1 -> conv1x1(w1)+b1 [WMMA] -> dwconv3x3+b2 -> SimpleGate
// Writes x1 packed [b][pix][32] bf16 and per-tile channel partial sums.
// ---------------------------------------------------------------------------
__global__ __launch_bounds__(256) void naf_stage1(
    const float* __restrict__ inp, const float* __restrict__ w1,
    const float* __restrict__ b1,  const float* __restrict__ w2,
    const float* __restrict__ b2,  const float* __restrict__ n1w,
    const float* __restrict__ n1b, __bf16* __restrict__ xws,
    float* __restrict__ partial)
{
    extern __shared__ float smem[];
    float* u    = smem;                  // [DW][USTRIDE] conv1 output (+halo)
    float* w1s  = u + DW * USTRIDE;      // 2048
    float* w2s  = w1s + DW * Cc;         // 576
    float* b1s  = w2s + DW * 9;          // 64
    float* b2s  = b1s + DW;              // 64
    float* n1ws = b2s + DW;              // 32
    float* n1bs = n1ws + Cc;             // 32
    float* pool = n1bs + Cc;             // 32

    const int t = threadIdx.x;
    const int b = blockIdx.z;
    const int tileX = blockIdx.x, tileY = blockIdx.y;
    const int wave = t >> 5, lane = t & 31;
    const int h = (lane >> 4) & 1, n = lane & 15;

    for (int i = t; i < DW * Cc; i += 256) w1s[i] = w1[b * DW * Cc + i];
    for (int i = t; i < DW * 9;  i += 256) w2s[i] = w2[b * DW * 9 + i];
    if (t < DW) { b1s[t] = b1[b * DW + t]; b2s[t] = b2[b * DW + t]; }
    if (t < Cc) { n1ws[t] = n1w[t]; n1bs[t] = n1b[t]; pool[t] = 0.f; }
    __syncthreads();
    // fold LN1 shift into bias:  b1'[o] = b1[o] + sum_c w1[o,c]*n1b[c]
    if (t < DW) {
        float acc = b1s[t];
        for (int c = 0; c < Cc; ++c) acc += w1s[t * Cc + c] * n1bs[c];
        b1s[t] = acc;
    }
    __syncthreads();
    // fold LN1 gain into A columns: w1'[o,c] = w1[o,c]*n1w[c]
    for (int i = t; i < DW * Cc; i += 256) w1s[i] *= n1ws[i & (Cc - 1)];
    __syncthreads();

    bf16x16 a1[4];
    f32x8  cb1[4];
#pragma unroll
    for (int m = 0; m < 4; ++m) {
        build_afrag(w1s, m, lane, a1[m]);
        cb1[m] = build_cbias(b1s, m, lane);
    }

    const size_t inpB = (size_t)b * Cc * HW;

    // ---- fill u = conv1(LN(inp))+b1 over tile+halo (zero pad at edges) ----
    for (int g = wave; g < NGROUP; g += 8) {
        const int hp = g * 16 + n;
        const int hy = hp / HALO_W, hx = hp - hy * HALO_W;
        const int gy = tileY * TILE_H - 1 + hy;
        const int gx = tileX * TILE_W - 1 + hx;
        const bool valid = (hp < HALO_PX) && (gy >= 0) && (gy < Hh) && (gx >= 0) && (gx < Ww);
        const float vmask = valid ? 1.f : 0.f;

        float xv[16];
        if (valid) {
            const float* ip = inp + inpB + (size_t)(16 * h) * HW + gy * Ww + gx;
#pragma unroll
            for (int k = 0; k < 16; ++k) xv[k] = ip[(size_t)k * HW];
        } else {
#pragma unroll
            for (int k = 0; k < 16; ++k) xv[k] = 0.f;
        }
        float sum = 0.f, sq = 0.f;
#pragma unroll
        for (int k = 0; k < 16; ++k) { sum += xv[k]; sq += xv[k] * xv[k]; }
        sum += __shfl_xor(sum, 16);   // lane pair (n, n+16) holds same pixel
        sq  += __shfl_xor(sq, 16);
        const float mu = sum * (1.f / 32.f);
        const float var = sq * (1.f / 32.f) - mu * mu;
        const float rstdm = rsqrtf(var + 1e-6f) * vmask;   // masked inv-std

        bf16x16 bx;   // B-fragment: lane pixel n, channels 16h..16h+15
#pragma unroll
        for (int k = 0; k < 16; ++k)
            bx[k] = (__bf16)((xv[k] - mu) * rstdm);   // affine folded into A/bias

        f32x8 acc[4];
#pragma unroll
        for (int m = 0; m < 4; ++m) acc[m] = wmma_bf16f32(a1[m], bx, cb1[m]);

        // invalid columns: conv part is 0, acc == bias -> *vmask gives exact 0
#pragma unroll
        for (int m = 0; m < 4; ++m)
#pragma unroll
            for (int r = 0; r < 8; ++r) {
                const int o = m * 16 + 8 * h + r;   // D: lane row = 8h+r
                u[o * USTRIDE + hp] = acc[m][r] * vmask;
            }
    }
    __syncthreads();

    // ---- depthwise 3x3 + SimpleGate + tile pooling, packed bf16 output ----
    const int tx = t & 31, ty = t >> 5;
    const int hbase = (ty + 1) * HALO_W + (tx + 1);
    const int gpix = (tileY * TILE_H + ty) * Ww + tileX * TILE_W + tx;
    __bf16* xp = xws + ((size_t)b * HW + gpix) * Cc;   // [pix][32] packed
#pragma unroll 1
    for (int cc = 0; cc < 4; ++cc) {
        float xg8[8];
#pragma unroll
        for (int j = 0; j < 8; ++j) {
            const int c = cc * 8 + j;
            const float* u1 = u + c * USTRIDE;
            const float* u2 = u + (c + Cc) * USTRIDE;
            const float* k1 = w2s + c * 9;
            const float* k2 = w2s + (c + Cc) * 9;
            float acc1 = 0.f, acc2 = 0.f;
#pragma unroll
            for (int dy = 0; dy < 3; ++dy)
#pragma unroll
                for (int dx = 0; dx < 3; ++dx) {
                    const int off = hbase + (dy - 1) * HALO_W + (dx - 1);
                    acc1 += k1[dy * 3 + dx] * u1[off];
                    acc2 += k2[dy * 3 + dx] * u2[off];
                }
            const float xg = (acc1 + b2s[c]) * (acc2 + b2s[c + Cc]);  // SimpleGate
            xg8[j] = xg;
            float ps = xg;
            ps += __shfl_xor(ps, 16); ps += __shfl_xor(ps, 8);
            ps += __shfl_xor(ps, 4);  ps += __shfl_xor(ps, 2);
            ps += __shfl_xor(ps, 1);
            if (lane == 0) atomicAdd(&pool[c], ps);
        }
        bf16x8 pk;
#pragma unroll
        for (int j = 0; j < 8; ++j) pk[j] = (__bf16)xg8[j];
        *(bf16x8*)(xp + cc * 8) = pk;                    // 16B packed store
    }
    __syncthreads();
    if (t < Cc) {
        const int tile = tileY * gridDim.x + tileX;
        partial[((size_t)b * TILES_PER_B + tile) * Cc + t] = pool[t];
    }
}

// ---------------------------------------------------------------------------
// SE: reduce tile partials -> pooled mean -> sca matmul + bias -> s[b][32]
// ---------------------------------------------------------------------------
__global__ __launch_bounds__(32) void naf_se(
    const float* __restrict__ partial, const float* __restrict__ scaw,
    const float* __restrict__ scab, float* __restrict__ sfin)
{
    __shared__ float pooled[Cc];
    const int b = blockIdx.x, t = threadIdx.x;
    float s = 0.f;
    for (int i = 0; i < TILES_PER_B; ++i) s += partial[((size_t)b * TILES_PER_B + i) * Cc + t];
    pooled[t] = s * (1.f / (float)HW);
    __syncthreads();
    float acc = scab[t];
    for (int c = 0; c < Cc; ++c) acc += pooled[c] * scaw[t * Cc + c];
    sfin[b * Cc + t] = acc;
}

// ---------------------------------------------------------------------------
// Stage 2: x1*s -> conv(w3)+b3 -> y = inp + x*beta -> LN2 -> conv(w4)+b4
//          -> SG -> conv(w5)+b5 -> out = y + x*gamma.  Fully fused.
// SE scale folded into w3 columns; LN2 affine folded into w4 columns + bias.
// ---------------------------------------------------------------------------
__global__ __launch_bounds__(256) void naf_stage2(
    const float* __restrict__ inp, const __bf16* __restrict__ xws,
    const float* __restrict__ sfin,
    const float* __restrict__ w3, const float* __restrict__ b3,
    const float* __restrict__ w4, const float* __restrict__ b4,
    const float* __restrict__ w5, const float* __restrict__ b5,
    const float* __restrict__ n2w, const float* __restrict__ n2b,
    const float* __restrict__ beta, const float* __restrict__ gamma,
    float* __restrict__ out)
{
    __shared__ float w3s[Cc * Cc], w4s[DW * Cc], w5s[Cc * Cc];
    __shared__ float b3s[Cc], b4s[DW], b5s[Cc];
    __shared__ float n2ws[Cc], n2bs[Cc], ss[Cc], bets[Cc], gams[Cc];

    const int t = threadIdx.x, b = blockIdx.y, chunk = blockIdx.x;
    const int wave = t >> 5, lane = t & 31;
    const int h = (lane >> 4) & 1, n = lane & 15;

    for (int i = t; i < Cc * Cc; i += 256) { w3s[i] = w3[b * Cc * Cc + i]; w5s[i] = w5[b * Cc * Cc + i]; }
    for (int i = t; i < DW * Cc; i += 256) w4s[i] = w4[b * DW * Cc + i];
    if (t < DW) b4s[t] = b4[b * DW + t];
    if (t < Cc) {
        b3s[t] = b3[b * Cc + t]; b5s[t] = b5[b * Cc + t];
        n2ws[t] = n2w[t]; n2bs[t] = n2b[t];
        ss[t] = sfin[b * Cc + t]; bets[t] = beta[t]; gams[t] = gamma[t];
    }
    __syncthreads();
    // fold LN2 shift into bias:  b4'[o] = b4[o] + sum_c w4[o,c]*n2b[c]
    if (t < DW) {
        float acc = b4s[t];
        for (int c = 0; c < Cc; ++c) acc += w4s[t * Cc + c] * n2bs[c];
        b4s[t] = acc;
    }
    __syncthreads();
    // fold SE scale into w3 columns; LN2 gain into w4 columns
    for (int i = t; i < Cc * Cc; i += 256) w3s[i] *= ss[i & (Cc - 1)];
    for (int i = t; i < DW * Cc; i += 256) w4s[i] *= n2ws[i & (Cc - 1)];
    __syncthreads();

    bf16x16 a3[2], a4[4], a5[2];
    f32x8  cb3[2], cb4[4], cb5[2];
#pragma unroll
    for (int m = 0; m < 2; ++m) {
        build_afrag(w3s, m, lane, a3[m]);  cb3[m] = build_cbias(b3s, m, lane);
        build_afrag(w5s, m, lane, a5[m]);  cb5[m] = build_cbias(b5s, m, lane);
    }
#pragma unroll
    for (int m = 0; m < 4; ++m) {
        build_afrag(w4s, m, lane, a4[m]);  cb4[m] = build_cbias(b4s, m, lane);
    }

    // per-lane scale/shift constants for this lane's D rows
    float betv0[8], betv1[8], gamv0[8], gamv1[8];
#pragma unroll
    for (int r = 0; r < 8; ++r) {
        const int c0 = 8 * h + r, c1 = 16 + 8 * h + r;
        betv0[r] = bets[c0]; betv1[r] = bets[c1];
        gamv0[r] = gams[c0]; gamv1[r] = gams[c1];
    }

    const size_t base = (size_t)b * Cc * HW;

    for (int it = 0; it < 16; ++it) {
        const int p = chunk * 2048 + (it * 8 + wave) * 16 + n;

        // --- conv1x1(w3*s)+b3: B-fragment is a direct 32B packed-bf16 load ---
        const bf16x16 bx = *(const bf16x16*)(xws + ((size_t)b * HW + p) * Cc + 16 * h);
        const f32x8 t0 = wmma_bf16f32(a3[0], bx, cb3[0]);
        const f32x8 t1 = wmma_bf16f32(a3[1], bx, cb3[1]);

        // --- residual y = inp + x*beta (D-layout: ch 8h+r / 16+8h+r) ---
        float y0[8], y1[8];
#pragma unroll
        for (int r = 0; r < 8; ++r) {
            const int c0 = 8 * h + r, c1 = 16 + 8 * h + r;
            y0[r] = inp[base + (size_t)c0 * HW + p] + t0[r] * betv0[r];
            y1[r] = inp[base + (size_t)c1 * HW + p] + t1[r] * betv1[r];
        }

        // --- D-layout -> B-layout (lane holds ch 16h..16h+15) via xor-16 swap ---
        float yB[16];
#pragma unroll
        for (int r = 0; r < 8; ++r) {
            const float sA = __shfl_xor(y1[r], 16);
            const float sB = __shfl_xor(y0[r], 16);
            yB[r]     = h ? sA : y0[r];
            yB[r + 8] = h ? y1[r] : sB;
        }

        // --- LN2 (affine folded into w4/b4) ---
        float sum = 0.f, sq = 0.f;
#pragma unroll
        for (int k = 0; k < 16; ++k) { sum += yB[k]; sq += yB[k] * yB[k]; }
        sum += __shfl_xor(sum, 16);
        sq  += __shfl_xor(sq, 16);
        const float mu = sum * (1.f / 32.f);
        const float rstd = rsqrtf(sq * (1.f / 32.f) - mu * mu + 1e-6f);
        bf16x16 bn;
#pragma unroll
        for (int k = 0; k < 16; ++k)
            bn[k] = (__bf16)((yB[k] - mu) * rstd);

        // --- conv1x1(w4*n2w)+b4': 64 outputs, 4 M-tiles ---
        const f32x8 v0 = wmma_bf16f32(a4[0], bn, cb4[0]);
        const f32x8 v1 = wmma_bf16f32(a4[1], bn, cb4[1]);
        const f32x8 v2 = wmma_bf16f32(a4[2], bn, cb4[2]);
        const f32x8 v3 = wmma_bf16f32(a4[3], bn, cb4[3]);

        // --- SimpleGate: ch c = v[c]*v[c+32]  (lane-local: m0*m2, m1*m3) ---
        float sg0[8], sg1[8];
#pragma unroll
        for (int r = 0; r < 8; ++r) {
            sg0[r] = v0[r] * v2[r];
            sg1[r] = v1[r] * v3[r];
        }

        // --- D->B transform + conv1x1(w5)+b5 ---
        bf16x16 bg;
#pragma unroll
        for (int r = 0; r < 8; ++r) {
            const float sA = __shfl_xor(sg1[r], 16);
            const float sB = __shfl_xor(sg0[r], 16);
            bg[r]     = (__bf16)(h ? sA : sg0[r]);
            bg[r + 8] = (__bf16)(h ? sg1[r] : sB);
        }
        const f32x8 o0 = wmma_bf16f32(a5[0], bg, cb5[0]);
        const f32x8 o1 = wmma_bf16f32(a5[1], bg, cb5[1]);

        // --- out = y + x*gamma ---
#pragma unroll
        for (int r = 0; r < 8; ++r) {
            const int c0 = 8 * h + r, c1 = 16 + 8 * h + r;
            out[base + (size_t)c0 * HW + p] = y0[r] + o0[r] * gamv0[r];
            out[base + (size_t)c1 * HW + p] = y1[r] + o1[r] * gamv1[r];
        }
    }
}

// ---------------------------------------------------------------------------
extern "C" void kernel_launch(void* const* d_in, const int* in_sizes, int n_in,
                              void* d_out, int out_size, void* d_ws, size_t ws_size,
                              hipStream_t stream) {
    const float* inp  = (const float*)d_in[0];
    const float* w1   = (const float*)d_in[1];
    const float* b1   = (const float*)d_in[2];
    const float* w2   = (const float*)d_in[3];
    const float* b2   = (const float*)d_in[4];
    const float* w3   = (const float*)d_in[5];
    const float* b3   = (const float*)d_in[6];
    const float* w4   = (const float*)d_in[7];
    const float* b4   = (const float*)d_in[8];
    const float* w5   = (const float*)d_in[9];
    const float* b5   = (const float*)d_in[10];
    const float* n1w  = (const float*)d_in[11];
    const float* n1b  = (const float*)d_in[12];
    const float* n2w  = (const float*)d_in[13];
    const float* n2b  = (const float*)d_in[14];
    const float* scaw = (const float*)d_in[15];
    const float* scab = (const float*)d_in[16];
    const float* beta = (const float*)d_in[17];
    const float* gamma= (const float*)d_in[18];

    // workspace: x1 packed [b][pix][32] bf16 | tile partial sums | s_final
    __bf16* xws    = (__bf16*)d_ws;
    float*  partial= (float*)((char*)d_ws + (size_t)Bn * HW * Cc * sizeof(__bf16));
    float*  sfin   = partial + (size_t)Bn * TILES_PER_B * Cc;

    const size_t lds1 = (size_t)(DW * USTRIDE + DW * Cc + DW * 9 + DW + DW + Cc + Cc + Cc) * sizeof(float);

    naf_stage1<<<dim3(Ww / TILE_W, Hh / TILE_H, Bn), 256, lds1, stream>>>(
        inp, w1, b1, w2, b2, n1w, n1b, xws, partial);
    naf_se<<<dim3(Bn), 32, 0, stream>>>(partial, scaw, scab, sfin);
    naf_stage2<<<dim3(HW / 2048, Bn), 256, 0, stream>>>(
        inp, xws, sfin, w3, b3, w4, b4, w5, b5, n2w, n2b, beta, gamma, (float*)d_out);
}
